// OfficeLSTM_42880953483456
// MI455X (gfx1250) — matrix operand order repaired and checked by reading, not compiled, not run on hardware
//
#include <hip/hip_runtime.h>
#include <hip/hip_bf16.h>
#include <cstdint>

// ---------------------------------------------------------------------------
// Persistent batch-tiled LSTM for MI455X (gfx1250, wave32, WMMA).
//   grid = 32 workgroups x 512 threads (16 waves). Each WG owns 16 batch rows
//   and runs all 256 timesteps, c-state in registers, h in padded LDS.
//   Matmuls via v_wmma_f32_16x16x32_f16 (fp32 accumulate, inline-0 C on the
//   first K-chunk). Weights stream from L2 (pre-transposed fp16, 3 MB <<
//   192 MB L2) through address_space(1) pointers -> global_load_b128.
// ---------------------------------------------------------------------------

typedef __attribute__((ext_vector_type(16))) _Float16 v16h;
typedef __attribute__((ext_vector_type(8)))  _Float16 v8h;
typedef __attribute__((ext_vector_type(8)))  float    v8f;

#define GAS __attribute__((address_space(1)))
typedef GAS const _Float16* gch_ptr;

#define B_SZ 512
#define T_SZ 256
#define D_SZ 256
#define U_SZ 512
#define G4   2048        // 4*U

#define BT      16       // batch rows per workgroup
#define NWAVES  16
#define THREADS (NWAVES * 32)
#define SA 264           // padded LDS row stride for x tile (256 + 8) -> bank-spread
#define SH 520           // padded LDS row stride for h    (512 + 8) -> bank-spread

__device__ __forceinline__ float fast_sig(float x) {
  x = fminf(30.0f, fmaxf(-30.0f, x));
  float e = __expf(x);
  return e * __builtin_amdgcn_rcpf(1.0f + e);
}
__device__ __forceinline__ float fast_tanh(float x) {
  x = fminf(15.0f, fmaxf(-15.0f, x));
  float e = __expf(2.0f * x);
  return (e - 1.0f) * __builtin_amdgcn_rcpf(e + 1.0f);
}

// A fragment (16x32 f16): lane = row (M) in 0..15; lane group selects K halves.
// Lanes 0-15: K = kb+0..15 ; lanes 16-31: K = kb+16..31.
// Two aligned 16B LDS loads per lane.
__device__ __forceinline__ v16h load_a_frag(const _Float16* base, int stride,
                                            int row, int hi, int kb) {
  const _Float16* p = base + row * stride + kb + 16 * hi;
  v8h lo = *(const v8h*)(p);
  v8h hv = *(const v8h*)(p + 8);
  v16h a;
#pragma unroll
  for (int i = 0; i < 8; ++i) { a[i] = lo[i]; a[8 + i] = hv[i]; }
  return a;
}

// B fragment (32x16 f16): lane = column N; lanes 0-15 hold K=kb..kb+15,
// lanes 16-31 hold K=kb+16..kb+31 (packed 2/VGPR in K order).
// Weights are pre-transposed to [N][K] so this is ONE contiguous 32B
// global load (address_space(1) guarantees the GLOBAL encoding).
__device__ __forceinline__ v16h load_b_frag(gch_ptr Bw, int Kstride,
                                            int n, int kb, int hi) {
  return *(GAS const v16h*)(Bw + (size_t)n * Kstride + kb + 16 * hi);
}

// ---------------------------------------------------------------------------
// One-time weight conversion: fp32 row-major [K][4U] -> fp16 B-layout [4U][K].
// ---------------------------------------------------------------------------
__global__ void convert_weights(const float* __restrict__ Wx,
                                const float* __restrict__ Wh,
                                _Float16* __restrict__ Bx,
                                _Float16* __restrict__ Bh) {
  const int stride = gridDim.x * blockDim.x;
  const int tid = blockIdx.x * blockDim.x + threadIdx.x;
  const int totalx = G4 * D_SZ;
  for (int i = tid; i < totalx; i += stride) {
    int n = i / D_SZ, k = i % D_SZ;
    Bx[i] = (_Float16)Wx[(size_t)k * G4 + n];
  }
  const int totalh = G4 * U_SZ;
  for (int i = tid; i < totalh; i += stride) {
    int n = i / U_SZ, k = i % U_SZ;
    Bh[i] = (_Float16)Wh[(size_t)k * G4 + n];
  }
}

// ---------------------------------------------------------------------------
// Persistent LSTM + MLP head.
// Wave w owns u in [32w, 32w+32) across ALL FOUR gates -> gate math is
// wave-local on the WMMA C/D register layout (VGPR r, lane l: row=r+8*(l>>4),
// col=l&15).
// ---------------------------------------------------------------------------
__global__ void __launch_bounds__(THREADS)
lstm_persist(const float* __restrict__ x,  const float* __restrict__ bias,
             const float* __restrict__ W1, const float* __restrict__ b1,
             const float* __restrict__ W2, const float* __restrict__ b2,
             const _Float16* __restrict__ Bx, const _Float16* __restrict__ Bh,
             float* __restrict__ out) {
  __shared__ __align__(16) _Float16 s_x[2][BT * SA];  // double-buffered x tile
  __shared__ __align__(16) _Float16 s_h[BT * SH];     // h_t, fp16
  __shared__ float s_hidden[BT * 128];
  __shared__ float s_logits[BT * 10];

  const int tid   = threadIdx.x;
  const int lane  = tid & 31;
  const int w     = tid >> 5;          // wave id 0..15
  const int hi    = lane >> 4;         // lane half
  const int row16 = lane & 15;
  const int ubase = w * 32;            // this wave's u-slice
  const int row0  = blockIdx.x * BT;   // batch base row

  // staging indices: BT*64 float4 = exactly 2 per thread
  const int i0 = tid, i1 = tid + THREADS;
  const int r0s = i0 >> 6, k0s = (i0 & 63) * 4;
  const int r1s = i1 >> 6, k1s = (i1 & 63) * 4;

  for (int i = tid; i < BT * SH; i += THREADS) s_h[i] = (_Float16)0.0f;

  // bias for this wave's 8 output tiles: one scalar per lane, preloaded into
  // registers BEFORE the time loop (barrier fences block LICM otherwise)
  float barr[4][2];
#pragma unroll
  for (int g = 0; g < 4; ++g)
#pragma unroll
    for (int j = 0; j < 2; ++j)
      barr[g][j] = bias[g * U_SZ + ubase + j * 16 + row16];

  v8f c_state[2];
#pragma unroll
  for (int j = 0; j < 2; ++j)
#pragma unroll
    for (int r = 0; r < 8; ++r) c_state[j][r] = 0.0f;

  // --- stage x_0 into buffer 0 ---
  {
    float4 v0 = ((const float4*)(x + ((size_t)(row0 + r0s)) * T_SZ * D_SZ))[k0s >> 2];
    float4 v1 = ((const float4*)(x + ((size_t)(row0 + r1s)) * T_SZ * D_SZ))[k1s >> 2];
    s_x[0][r0s * SA + k0s + 0] = (_Float16)v0.x;
    s_x[0][r0s * SA + k0s + 1] = (_Float16)v0.y;
    s_x[0][r0s * SA + k0s + 2] = (_Float16)v0.z;
    s_x[0][r0s * SA + k0s + 3] = (_Float16)v0.w;
    s_x[0][r1s * SA + k1s + 0] = (_Float16)v1.x;
    s_x[0][r1s * SA + k1s + 1] = (_Float16)v1.y;
    s_x[0][r1s * SA + k1s + 2] = (_Float16)v1.z;
    s_x[0][r1s * SA + k1s + 3] = (_Float16)v1.w;
  }
  __syncthreads();

  // Global (AS1) weight pointers, laundered each timestep through an empty
  // asm so the compiler cannot hoist (and then spill) the L2-resident weight
  // fragments out of the 256-iteration time loop.
  gch_ptr bx_t = (gch_ptr)Bx;
  gch_ptr bh_t = (gch_ptr)Bh;

  for (int t = 0; t < T_SZ; ++t) {
    asm volatile("" : "+s"(bx_t), "+s"(bh_t));
    const _Float16* cur = s_x[t & 1];

    // --- early-issue next timestep's x loads (latency hidden under WMMAs) ---
    float4 vx0, vx1;
    const bool have_next = (t + 1 < T_SZ);   // uniform branch: EXEC stays full
    if (have_next) {
      vx0 = ((const float4*)(x + (((size_t)(row0 + r0s)) * T_SZ + (t + 1)) * D_SZ))[k0s >> 2];
      vx1 = ((const float4*)(x + (((size_t)(row0 + r1s)) * T_SZ + (t + 1)) * D_SZ))[k1s >> 2];
    }

    // --- z = x_t @ Wx, first K-chunk peeled: C = inline 0 (no mov splats) ---
    v8f acc[4][2];
    {
      v16h a = load_a_frag(cur, SA, row16, hi, 0);
#pragma unroll
      for (int g = 0; g < 4; ++g)
#pragma unroll
        for (int j = 0; j < 2; ++j) {
          int n = g * U_SZ + ubase + j * 16;
          v16h bf = load_b_frag(bx_t, D_SZ, n, 0, hi);
          v8f z = {};  // folds to the WMMA inline-0 SRC2 operand
          acc[g][j] = __builtin_amdgcn_wmma_f32_16x16x32_f16(
              false, a, false, bf, (short)0, z, false, false);
        }
    }
    for (int kb = 32; kb < D_SZ; kb += 32) {
      v16h a = load_a_frag(cur, SA, row16, hi, kb);
#pragma unroll
      for (int g = 0; g < 4; ++g)
#pragma unroll
        for (int j = 0; j < 2; ++j) {
          int n = g * U_SZ + ubase + j * 16;
          v16h bf = load_b_frag(bx_t, D_SZ, n, kb, hi);
          acc[g][j] = __builtin_amdgcn_wmma_f32_16x16x32_f16(
              false, a, false, bf, (short)0, acc[g][j], false, false);
        }
    }

    // --- z += h_{t-1} @ Wh   (K = 512) ---
    for (int kb = 0; kb < U_SZ; kb += 32) {
      v16h a = load_a_frag(s_h, SH, row16, hi, kb);
      if (kb + 32 < U_SZ)  // stream next weight chunk toward the caches
        __builtin_prefetch(Bh + (size_t)ubase * U_SZ + kb + 32, 0, 0);
#pragma unroll
      for (int g = 0; g < 4; ++g)
#pragma unroll
        for (int j = 0; j < 2; ++j) {
          int n = g * U_SZ + ubase + j * 16;
          v16h bf = load_b_frag(bh_t, U_SZ, n, kb, hi);
          acc[g][j] = __builtin_amdgcn_wmma_f32_16x16x32_f16(
              false, a, false, bf, (short)0, acc[g][j], false, false);
        }
    }

    // --- gate math BEFORE the barrier: register-only (overlaps with other
    //     waves still in their K-loops). keras gate order i,f,g,o. ---
    _Float16 hv16[2][8];
#pragma unroll
    for (int j = 0; j < 2; ++j) {
#pragma unroll
      for (int r = 0; r < 8; ++r) {
        float iv = fast_sig (acc[0][j][r] + barr[0][j]);
        float fv = fast_sig (acc[1][j][r] + barr[1][j]);
        float gv = fast_tanh(acc[2][j][r] + barr[2][j]);
        float ov = fast_sig (acc[3][j][r] + barr[3][j]);
        float cv = fv * c_state[j][r] + iv * gv;
        c_state[j][r] = cv;
        hv16[j][r] = (_Float16)(ov * fast_tanh(cv));
      }
    }

    // --- store next x tile into the other buffer (no reader until t+1) ---
    if (have_next) {
      _Float16* nxt = s_x[(t + 1) & 1];
      nxt[r0s * SA + k0s + 0] = (_Float16)vx0.x;
      nxt[r0s * SA + k0s + 1] = (_Float16)vx0.y;
      nxt[r0s * SA + k0s + 2] = (_Float16)vx0.z;
      nxt[r0s * SA + k0s + 3] = (_Float16)vx0.w;
      nxt[r1s * SA + k1s + 0] = (_Float16)vx1.x;
      nxt[r1s * SA + k1s + 1] = (_Float16)vx1.y;
      nxt[r1s * SA + k1s + 2] = (_Float16)vx1.z;
      nxt[r1s * SA + k1s + 3] = (_Float16)vx1.w;
    }
    __syncthreads();  // all reads of s_h / cur done; next-x staged

    // --- tiny critical section: just the h stores ---
#pragma unroll
    for (int j = 0; j < 2; ++j) {
      int col = ubase + j * 16 + row16;
#pragma unroll
      for (int r = 0; r < 8; ++r)
        s_h[(8 * hi + r) * SH + col] = hv16[j][r];
    }
    __syncthreads();  // new h visible before next iteration's reads
  }

  // --- hidden = relu(h @ W1 + b1) : [16,128] ---
  for (int i = tid; i < BT * 128; i += THREADS) {
    int r = i >> 7, jc = i & 127;
    float s = b1[jc];
    const _Float16* hrow = s_h + r * SH;
#pragma unroll 4
    for (int u = 0; u < U_SZ; ++u) s += (float)hrow[u] * W1[(size_t)u * 128 + jc];
    s_hidden[i] = fmaxf(s, 0.0f);
  }
  __syncthreads();

  // --- logits = hidden @ W2 + b2 : [16,10] ---
  for (int i = tid; i < BT * 10; i += THREADS) {
    int r = i / 10, c = i % 10;
    float s = b2[c];
#pragma unroll 4
    for (int u = 0; u < 128; ++u) s += s_hidden[r * 128 + u] * W2[u * 10 + c];
    s_logits[i] = s;
  }
  __syncthreads();

  // --- softmax per batch row ---
  if (tid < BT) {
    int r = tid;
    float m = -3.0e38f;
#pragma unroll
    for (int c = 0; c < 10; ++c) m = fmaxf(m, s_logits[r * 10 + c]);
    float e[10], sum = 0.0f;
#pragma unroll
    for (int c = 0; c < 10; ++c) {
      e[c] = __expf(s_logits[r * 10 + c] - m);
      sum += e[c];
    }
    float inv = 1.0f / sum;
#pragma unroll
    for (int c = 0; c < 10; ++c)
      out[(size_t)(row0 + r) * 10 + c] = e[c] * inv;
  }
}

extern "C" void kernel_launch(void* const* d_in, const int* in_sizes, int n_in,
                              void* d_out, int out_size, void* d_ws, size_t ws_size,
                              hipStream_t stream) {
  const float* x  = (const float*)d_in[0];  // [B,T,D]
  const float* Wx = (const float*)d_in[1];  // [D,4U]
  const float* Wh = (const float*)d_in[2];  // [U,4U]
  const float* b  = (const float*)d_in[3];  // [4U]
  const float* W1 = (const float*)d_in[4];  // [U,128]
  const float* b1 = (const float*)d_in[5];  // [128]
  const float* W2 = (const float*)d_in[6];  // [128,10]
  const float* b2 = (const float*)d_in[7];  // [10]
  float* out = (float*)d_out;               // [B,10]

  // workspace: fp16 weights in B-fragment layout [N][K]
  _Float16* Bx = (_Float16*)d_ws;                                   // 1 MB
  _Float16* Bh = (_Float16*)((char*)d_ws +
                             (size_t)G4 * D_SZ * sizeof(_Float16)); // 2 MB

  convert_weights<<<512, 256, 0, stream>>>(Wx, Wh, Bx, Bh);
  lstm_persist<<<B_SZ / BT, THREADS, 0, stream>>>(x, b, W1, b1, W2, b2,
                                                  Bx, Bh, out);
}
